// TemporalGNNCell_3642132267417
// MI455X (gfx1250) — matrix operand order repaired and checked by reading, not compiled, not run on hardware
//
#include <hip/hip_runtime.h>
#include <hip/hip_bf16.h>
#include <math.h>

typedef __attribute__((ext_vector_type(16))) _Float16 v16h;
typedef __attribute__((ext_vector_type(8)))  float    v8f;

#define B_    2
#define N_    20000
#define C_    128
#define E_    320000
#define H_    4
#define COUT_ 64
#define HD_   256
#define ED_   16
#define BN_   (B_ * N_)            // 40000
#define ETOT_ (B_ * E_ + BN_)      // 680000

// ---------- helpers ----------
__device__ __forceinline__ unsigned f2ord(float f) {
    unsigned u = __float_as_uint(f);
    return (u >> 31) ? ~u : (u | 0x80000000u);
}
__device__ __forceinline__ float ord2f(unsigned u) {
    return (u >> 31) ? __uint_as_float(u ^ 0x80000000u) : __uint_as_float(~u);
}
__device__ __forceinline__ void edge_nodes(int e, const int* __restrict__ ei,
                                           int& s, int& d, int& orig) {
    if (e < B_ * E_) {
        int b = e / E_;
        orig = e - b * E_;
        s = ei[orig]      + b * N_;    // edge_index[0]
        d = ei[E_ + orig] + b * N_;    // edge_index[1]
    } else {
        s = d = e - B_ * E_;           // self loop
        orig = -1;
    }
}
__device__ __forceinline__ float sigmoidf(float x) { return 1.f / (1.f + __expf(-x)); }

// Fragment swizzle: element (row m / col n in tile, k) -> (lane, half-slot)
// A 16x32 f16: lane = m + 16*((k>>3)&1), hi = 8*(k>>4) + (k&7)   (k = k mod 32)
// B 32x16 f16: lane = n + 16*((k>>3)&1), hi = 8*(k>>4) + (k&7)   (k = k mod 32)
// We store BOTH in the "A-gather-free" packed form: [k0step][lane][hi] contiguous.
__device__ __forceinline__ void a_swizzle(int m, int k, int& k0s, int& lane, int& hi) {
    int kr = k & 31; k0s = k >> 5;
    lane = m + (((kr >> 3) & 1) << 4);
    hi   = ((kr >> 4) << 3) + (kr & 7);
}
__device__ __forceinline__ void b_swizzle(int n, int k, int& k0s, int& lane, int& hi) {
    int kr = k & 31; k0s = k >> 5;
    lane = n + (((kr >> 3) & 1) << 4);
    hi   = ((kr >> 4) << 3) + (kr & 7);
}

// ---------- init: softmax max (-inf encoded), sums, aggregate buffer ----------
__global__ void k_init(unsigned* __restrict__ m, float* __restrict__ ssum,
                       float* __restrict__ spatial) {
    size_t t = (size_t)blockIdx.x * blockDim.x + threadIdx.x;
    size_t stride = (size_t)gridDim.x * blockDim.x;
    for (size_t i = t; i < (size_t)BN_ * H_; i += stride) { m[i] = 0x007FFFFFu; ssum[i] = 0.f; }
    for (size_t i = t; i < (size_t)BN_ * HD_; i += stride) spatial[i] = 0.f;
}

// ---------- pre-swizzle lin_w (128x256) into B-fragment f16 layout ----------
// dst layout: [jTile 16][k0s 4][lane 32][hi 16] halves
__global__ __launch_bounds__(256)
void k_cvt_lin(const float* __restrict__ lin_w, _Float16* __restrict__ dst) {
    int idx = blockIdx.x * blockDim.x + threadIdx.x;
    if (idx >= C_ * HD_) return;
    int k = idx >> 8, j = idx & 255;           // lin_w[k*256 + j]
    int jT = j >> 4, n = j & 15;
    int k0s, lane, hi;
    b_swizzle(n, k, k0s, lane, hi);
    dst[(size_t)(((jT << 2) + k0s) << 9) + (lane << 4) + hi] = (_Float16)lin_w[idx];
}

// ---------- pre-swizzle w_ih gates i,g,o into B-fragment f16 layout ----------
// dst layout: [g = l*3+t, 9][jTile 16][k0s 8][lane 32][hi 16] halves
__global__ __launch_bounds__(256)
void k_cvt_wih(const float* __restrict__ w_ih, _Float16* __restrict__ dst) {
    int idx = blockIdx.x * blockDim.x + threadIdx.x;
    if (idx >= 9 * 65536) return;
    int g = idx >> 16;                          // l*3 + t
    int j = (idx >> 8) & 255;
    int k = idx & 255;
    int l = g / 3, t = g - 3 * l;
    const int off[3] = {0, 2 * HD_, 3 * HD_};   // i, g, o gate row offsets
    float v = w_ih[((size_t)l * 4 * HD_ + off[t] + j) * HD_ + k];
    int jT = j >> 4, n = j & 15;
    int k0s, lane, hi;
    b_swizzle(n, k, k0s, lane, hi);
    dst[(size_t)((((g << 4) + jT) << 3) + k0s) * 512 + (lane << 4) + hi] = (_Float16)v;
}

// ---------- xt = x @ lin_w  (40000x128 @ 128x256), WMMA f16 ----------
__global__ __launch_bounds__(512)
void k_gemm_xt(const float* __restrict__ x, const _Float16* __restrict__ wlin,
               float* __restrict__ xt) {
    __shared__ v16h sA[4 * 32];                 // 16x128 A tile, fragment layout, 4 KB
    const int lane  = threadIdx.x & 31;
    const int wave  = threadIdx.x >> 5;         // 16 waves -> N tiles of 16 = 256 cols
    const int mBase = blockIdx.x * 16;
    _Float16* sAh = (_Float16*)sA;
    for (int idx = threadIdx.x; idx < 16 * C_; idx += 512) {
        int m = idx >> 7, k = idx & 127;        // coalesced read of x
        int k0s, sl, hi;
        a_swizzle(m, k, k0s, sl, hi);
        sAh[(k0s << 9) + (sl << 4) + hi] = (_Float16)x[(size_t)(mBase + m) * C_ + k];
    }
    __syncthreads();
    v8f acc = {};
    const v16h* bw = (const v16h*)wlin;
#pragma unroll
    for (int k0s = 0; k0s < 4; ++k0s) {
        v16h a = sA[(k0s << 5) + lane];                          // ds_load 32B
        v16h b = bw[(((wave << 2) + k0s) << 5) + lane];          // coalesced 1KB/wave
        acc = __builtin_amdgcn_wmma_f32_16x16x32_f16(false, a, false, b, (short)0, acc,
                                                     false, false);
    }
    const int nCol = lane & 15;
    const int rOff = (lane >> 4) << 3;
#pragma unroll
    for (int r = 0; r < 8; ++r)
        xt[(size_t)(mBase + r + rOff) * HD_ + wave * 16 + nCol] = acc[r];
}

// ---------- per-node attention scores a_src/a_dst (Bn x H) ----------
__global__ __launch_bounds__(256)
void k_node_attn(const float* __restrict__ xt, const float* __restrict__ att_src,
                 const float* __restrict__ att_dst, float* __restrict__ a_src,
                 float* __restrict__ a_dst) {
    const int lane = threadIdx.x & 31;
    const int node = blockIdx.x * 8 + (threadIdx.x >> 5);
    if (node >= BN_) return;
    float ps[H_], pd[H_];
#pragma unroll
    for (int h = 0; h < H_; ++h) { ps[h] = 0.f; pd[h] = 0.f; }
#pragma unroll
    for (int k = 0; k < 8; ++k) {               // cols lane+32k, head = k/2 (uniform per k)
        int c = lane + 32 * k;
        float v = xt[(size_t)node * HD_ + c];
        ps[k >> 1] += v * att_src[c];
        pd[k >> 1] += v * att_dst[c];
    }
#pragma unroll
    for (int h = 0; h < H_; ++h)
#pragma unroll
        for (int off = 16; off; off >>= 1) {
            ps[h] += __shfl_xor(ps[h], off, 32);
            pd[h] += __shfl_xor(pd[h], off, 32);
        }
    if (lane == 0)
#pragma unroll
        for (int h = 0; h < H_; ++h) {
            a_src[node * H_ + h] = ps[h];
            a_dst[node * H_ + h] = pd[h];
        }
}

// ---------- fold lin_edge_w & att_edge into 16x4 ----------
__global__ void k_edge_w(const float* __restrict__ lin_edge_w,
                         const float* __restrict__ att_edge, float* __restrict__ w_es) {
    int t = threadIdx.x;                        // 64 threads
    int d = t & 15, h = t >> 4;
    float s = 0.f;
    for (int c = 0; c < COUT_; ++c)
        s += lin_edge_w[d * HD_ + h * COUT_ + c] * att_edge[h * COUT_ + c];
    w_es[d * H_ + h] = s;
}

// ---------- per-edge edge-feature scores es (E x H) ----------
__global__ __launch_bounds__(256)
void k_edge_score(const float* __restrict__ edge_attr, const float* __restrict__ w_es,
                  float* __restrict__ es) {
    int e = blockIdx.x * blockDim.x + threadIdx.x;
    if (e >= E_) return;
    float ea[ED_];
    const float4* p = (const float4*)(edge_attr + (size_t)e * ED_);
#pragma unroll
    for (int q = 0; q < 4; ++q) {
        float4 v = p[q];
        ea[q * 4] = v.x; ea[q * 4 + 1] = v.y; ea[q * 4 + 2] = v.z; ea[q * 4 + 3] = v.w;
    }
#pragma unroll
    for (int h = 0; h < H_; ++h) {
        float s = 0.f;
#pragma unroll
        for (int d = 0; d < ED_; ++d) s += ea[d] * w_es[d * H_ + h];
        es[(size_t)e * H_ + h] = s;
    }
}

// ---------- alpha = leaky_relu(a_src+a_dst+es); segment max ----------
__global__ __launch_bounds__(256)
void k_alpha_max(const int* __restrict__ ei, const float* __restrict__ a_src,
                 const float* __restrict__ a_dst, const float* __restrict__ es,
                 float* __restrict__ raw, unsigned* __restrict__ m) {
    int e = blockIdx.x * blockDim.x + threadIdx.x;
    if (e >= ETOT_) return;
    int s, d, orig;
    edge_nodes(e, ei, s, d, orig);
#pragma unroll
    for (int h = 0; h < H_; ++h) {
        float a = a_src[s * H_ + h] + a_dst[d * H_ + h] +
                  (orig >= 0 ? es[(size_t)orig * H_ + h] : 0.f);
        a = (a > 0.f) ? a : 0.2f * a;
        raw[(size_t)e * H_ + h] = a;
        atomicMax(&m[d * H_ + h], f2ord(a));
    }
}

// ---------- alpha = exp(raw - max); segment sum ----------
__global__ __launch_bounds__(256)
void k_exp_sum(const int* __restrict__ ei, const unsigned* __restrict__ m,
               float* __restrict__ alpha, float* __restrict__ ssum) {
    int e = blockIdx.x * blockDim.x + threadIdx.x;
    if (e >= ETOT_) return;
    int s, d, orig;
    edge_nodes(e, ei, s, d, orig);
#pragma unroll
    for (int h = 0; h < H_; ++h) {
        float a = __expf(alpha[(size_t)e * H_ + h] - ord2f(m[d * H_ + h]));
        alpha[(size_t)e * H_ + h] = a;
        atomicAdd(&ssum[d * H_ + h], a);
    }
}

// ---------- normalize + scatter-add aggregate (dominant memory phase) ----------
__global__ __launch_bounds__(256)
void k_aggregate(const int* __restrict__ ei, const float* __restrict__ xt,
                 const float* __restrict__ alpha, const float* __restrict__ ssum,
                 float* __restrict__ spatial) {
    size_t t = (size_t)blockIdx.x * blockDim.x + threadIdx.x;
    if (t >= (size_t)ETOT_ * 64) return;
    int e = (int)(t >> 6);
    int q = (int)(t & 63);                      // 4 floats each; head = (q*4)/64 = q/16
    int s, d, orig;
    edge_nodes(e, ei, s, d, orig);
    int h = q >> 4;
    float an = alpha[(size_t)e * H_ + h] / (ssum[d * H_ + h] + 1e-16f);
    const float4 v = *(const float4*)(xt + (size_t)s * HD_ + q * 4);
    float* o = spatial + (size_t)d * HD_ + q * 4;
    atomicAdd(o + 0, v.x * an);
    atomicAdd(o + 1, v.y * an);
    atomicAdd(o + 2, v.z * an);
    atomicAdd(o + 3, v.w * an);
}

// ---------- add gat_bias ----------
__global__ void k_bias(float* __restrict__ spatial, const float* __restrict__ gat_bias) {
    size_t t = (size_t)blockIdx.x * blockDim.x + threadIdx.x;
    if (t >= (size_t)BN_ * HD_) return;
    spatial[t] += gat_bias[t & (HD_ - 1)];
}

// ---------- LSTM layer (h0=c0=0 => f gate dead, w_hh unused): WMMA ----------
__global__ __launch_bounds__(512)
void k_lstm(const float* __restrict__ h_in, const _Float16* __restrict__ wsw,
            const float* __restrict__ bih, const float* __restrict__ bhh,
            float* __restrict__ h_out, float* __restrict__ c_out) {
    __shared__ v16h sA[8 * 32];                 // 16x256 A tile, fragment layout, 8 KB
    const int lane  = threadIdx.x & 31;
    const int wave  = threadIdx.x >> 5;         // jTile = wave
    const int mBase = blockIdx.x * 16;
    _Float16* sAh = (_Float16*)sA;
    for (int idx = threadIdx.x; idx < 16 * HD_; idx += 512) {
        int m = idx >> 8, k = idx & 255;        // coalesced read of h_in
        int k0s, sl, hi;
        a_swizzle(m, k, k0s, sl, hi);
        sAh[(k0s << 9) + (sl << 4) + hi] = (_Float16)h_in[(size_t)(mBase + m) * HD_ + k];
    }
    __syncthreads();
    v8f av[3] = {{}, {}, {}};
    const v16h* bw = (const v16h*)wsw;
#pragma unroll
    for (int k0s = 0; k0s < 8; ++k0s) {
        v16h a = sA[(k0s << 5) + lane];
        if (k0s < 7)                            // global_prefetch_b8 on weight stream
            __builtin_prefetch(&bw[(((wave << 3) + k0s + 1) << 5) + lane], 0, 1);
#pragma unroll
        for (int t = 0; t < 3; ++t) {
            v16h b = bw[((((t << 4) + wave) << 3) + k0s) * 32 + lane];
            av[t] = __builtin_amdgcn_wmma_f32_16x16x32_f16(false, a, false, b, (short)0,
                                                           av[t], false, false);
        }
    }
    const int j  = wave * 16 + (lane & 15);
    const float bi = bih[j] + bhh[j];
    const float bg = bih[2 * HD_ + j] + bhh[2 * HD_ + j];
    const float bo = bih[3 * HD_ + j] + bhh[3 * HD_ + j];
    const int rOff = (lane >> 4) << 3;
#pragma unroll
    for (int r = 0; r < 8; ++r) {
        const size_t row = (size_t)(mBase + r + rOff);
        float c  = sigmoidf(av[0][r] + bi) * tanhf(av[1][r] + bg);
        float hv = sigmoidf(av[2][r] + bo) * tanhf(c);
        c_out[row * HD_ + j] = c;
        h_out[row * HD_ + j] = hv;
    }
}

// ---------- LayerNorm over HD=256, one wave32 per node ----------
__global__ __launch_bounds__(256)
void k_layernorm(const float* __restrict__ h, const float* __restrict__ g,
                 const float* __restrict__ b, float* __restrict__ out) {
    const int lane = threadIdx.x & 31;
    const int node = blockIdx.x * 8 + (threadIdx.x >> 5);
    if (node >= BN_) return;
    float v[8], s = 0.f, s2 = 0.f;
#pragma unroll
    for (int k = 0; k < 8; ++k) {
        v[k] = h[(size_t)node * HD_ + lane + 32 * k];
        s += v[k];
        s2 += v[k] * v[k];
    }
#pragma unroll
    for (int off = 16; off; off >>= 1) {
        s  += __shfl_xor(s, off, 32);
        s2 += __shfl_xor(s2, off, 32);
    }
    float mu  = s * (1.f / HD_);
    float var = s2 * (1.f / HD_) - mu * mu;
    float rs  = rsqrtf(var + 1e-5f);
#pragma unroll
    for (int k = 0; k < 8; ++k) {
        int c = lane + 32 * k;
        out[(size_t)node * HD_ + c] = (v[k] - mu) * rs * g[c] + b[c];
    }
}

extern "C" void kernel_launch(void* const* d_in, const int* in_sizes, int n_in,
                              void* d_out, int out_size, void* d_ws, size_t ws_size,
                              hipStream_t stream) {
    (void)in_sizes; (void)n_in; (void)out_size; (void)ws_size;
    const float* x          = (const float*)d_in[0];
    const int*   ei         = (const int*)d_in[1];
    const float* edge_attr  = (const float*)d_in[2];
    const float* lin_w      = (const float*)d_in[3];
    const float* att_src    = (const float*)d_in[4];
    const float* att_dst    = (const float*)d_in[5];
    const float* lin_edge_w = (const float*)d_in[6];
    const float* att_edge   = (const float*)d_in[7];
    const float* gat_bias   = (const float*)d_in[8];
    const float* w_ih       = (const float*)d_in[9];
    // d_in[10] w_hh: unused, h0 == 0
    const float* b_ih       = (const float*)d_in[11];
    const float* b_hh       = (const float*)d_in[12];
    const float* ln_g       = (const float*)d_in[13];
    const float* ln_b       = (const float*)d_in[14];
    float* out = (float*)d_out;

    // workspace carve-up (floats)
    float* wsf     = (float*)d_ws;
    float* xt      = wsf;  wsf += (size_t)BN_ * HD_;     // 10.24M
    float* spatial = wsf;  wsf += (size_t)BN_ * HD_;     // 10.24M
    float* a_src   = wsf;  wsf += (size_t)BN_ * H_;
    float* a_dst   = wsf;  wsf += (size_t)BN_ * H_;
    float* es      = wsf;  wsf += (size_t)E_ * H_;
    float* alpha   = wsf;  wsf += (size_t)ETOT_ * H_;
    float* ssum    = wsf;  wsf += (size_t)BN_ * H_;
    unsigned* mmax = (unsigned*)wsf; wsf += (size_t)BN_ * H_;
    float* w_es    = wsf;  wsf += 64;
    _Float16* wlin16 = (_Float16*)wsf; wsf += (size_t)(C_ * HD_) / 2;        // 64 KB
    _Float16* wih16  = (_Float16*)wsf; wsf += (size_t)(9 * 65536) / 2;       // 1.15 MB

    float* hs = out + (size_t)BN_ * HD_;                 // stacked h per layer
    float* cs = hs + 3 * (size_t)BN_ * HD_;              // stacked c per layer

    k_init<<<2048, 256, 0, stream>>>(mmax, ssum, spatial);
    k_cvt_lin<<<(C_ * HD_ + 255) / 256, 256, 0, stream>>>(lin_w, wlin16);
    k_cvt_wih<<<(9 * 65536 + 255) / 256, 256, 0, stream>>>(w_ih, wih16);
    k_gemm_xt<<<BN_ / 16, 512, 0, stream>>>(x, wlin16, xt);
    k_node_attn<<<BN_ / 8, 256, 0, stream>>>(xt, att_src, att_dst, a_src, a_dst);
    k_edge_w<<<1, 64, 0, stream>>>(lin_edge_w, att_edge, w_es);
    k_edge_score<<<(E_ + 255) / 256, 256, 0, stream>>>(edge_attr, w_es, es);
    k_alpha_max<<<(ETOT_ + 255) / 256, 256, 0, stream>>>(ei, a_src, a_dst, es, alpha, mmax);
    k_exp_sum<<<(ETOT_ + 255) / 256, 256, 0, stream>>>(ei, mmax, alpha, ssum);
    {
        long long t = (long long)ETOT_ * 64;
        k_aggregate<<<(unsigned)((t + 255) / 256), 256, 0, stream>>>(ei, xt, alpha, ssum,
                                                                     spatial);
    }
    k_bias<<<(unsigned)(((size_t)BN_ * HD_ + 255) / 256), 256, 0, stream>>>(spatial, gat_bias);

    const float* hin = spatial;
    for (int l = 0; l < 3; ++l) {
        float* hl = hs + (size_t)l * BN_ * HD_;
        float* cl = cs + (size_t)l * BN_ * HD_;
        k_lstm<<<BN_ / 16, 512, 0, stream>>>(hin, wih16 + (size_t)l * 3 * 16 * 8 * 512,
                                             b_ih + (size_t)l * 4 * HD_,
                                             b_hh + (size_t)l * 4 * HD_, hl, cl);
        hin = hl;
    }
    k_layernorm<<<BN_ / 8, 256, 0, stream>>>(hin, ln_g, ln_b, out);
}
